// MatryoshkaVSAEIso_32461362823169
// MI455X (gfx1250) — compile-verified, matrix operand order checked by reading, not run on hardware
//
#include <hip/hip_runtime.h>
#include <hip/hip_bf16.h>
#include <stdint.h>

#define ACT    2048
#define DICT   32768
#define BATCH  2048
#define KTOP   64
#define KTOTAL (KTOP * BATCH)   /* 131072 global top-k elements */

typedef __attribute__((ext_vector_type(16))) __bf16 v16bf;
typedef __attribute__((ext_vector_type(8)))  __bf16 v8bf;
typedef __attribute__((ext_vector_type(4)))  __bf16 v4bf;
typedef __attribute__((ext_vector_type(8)))  float  v8f;

// ---------------------------------------------------------------- tiling ----
constexpr int BM   = 128;  // batch rows per workgroup
constexpr int BN   = 64;   // output cols per workgroup
constexpr int BK   = 32;   // K slice (one bf16 WMMA depth)
constexpr int PADF = 4;    // f32 LDS pad: pitch 144B (16B-aligned)
constexpr int PADH = 8;    // bf16 LDS pad: pitch 80B (16B-aligned)

// ------------------------------------------------- CDNA5 async DMA helpers --
__device__ __forceinline__ unsigned lds_off(const void* p) {
  return (unsigned)(uintptr_t)p;   // generic LDS pointer: low 32 bits = offset
}
__device__ __forceinline__ void async_b128(unsigned lds, const void* g) {
  asm volatile("global_load_async_to_lds_b128 %0, %1, off"
               :: "v"(lds), "v"(g) : "memory");
}
__device__ __forceinline__ void wait_async0() {
  asm volatile("s_wait_asynccnt 0x0" ::: "memory");
}

// --------------------------------------------------- fragment gathers ------
// CDNA5 16-bit A/B layout: lane-half 0 holds K 0..7 (v0..3) and 16..23 (v4..7);
// lane-half 1 holds K 8..15 and 24..31.
__device__ __forceinline__ v16bf frag_bf16(const __bf16* rowp, int half) {
  v8bf lo = *(const v8bf*)(rowp + half * 8);        // K half*8 .. +7   (16B)
  v8bf hi = *(const v8bf*)(rowp + 16 + half * 8);   // K 16+half*8 ..   (16B)
  return __builtin_shufflevector(lo, hi, 0, 1, 2, 3, 4, 5, 6, 7,
                                 8, 9, 10, 11, 12, 13, 14, 15);
}
__device__ __forceinline__ v16bf frag_f32(const float* rowp, int half) {
  float4 q0 = *(const float4*)(rowp + half * 8 + 0);
  float4 q1 = *(const float4*)(rowp + half * 8 + 4);
  float4 q2 = *(const float4*)(rowp + 16 + half * 8 + 0);
  float4 q3 = *(const float4*)(rowp + 16 + half * 8 + 4);
  v16bf f;
  f[0]  = (__bf16)q0.x; f[1]  = (__bf16)q0.y; f[2]  = (__bf16)q0.z; f[3]  = (__bf16)q0.w;
  f[4]  = (__bf16)q1.x; f[5]  = (__bf16)q1.y; f[6]  = (__bf16)q1.z; f[7]  = (__bf16)q1.w;
  f[8]  = (__bf16)q2.x; f[9]  = (__bf16)q2.y; f[10] = (__bf16)q2.z; f[11] = (__bf16)q2.w;
  f[12] = (__bf16)q3.x; f[13] = (__bf16)q3.y; f[14] = (__bf16)q3.z; f[15] = (__bf16)q3.w;
  return f;
}

// ------------------------------------------------------------- encoder -----
// mu = x @ W_enc^T + b_enc ; lv = x @ W_var^T + b_var
__global__ __launch_bounds__(256)
void vsae_encoder_gemm(const float* __restrict__ x,
                       const float* __restrict__ Wenc,
                       const float* __restrict__ benc,
                       const float* __restrict__ Wvar,
                       const float* __restrict__ bvar,
                       float* __restrict__ mu,
                       float* __restrict__ lv)
{
  __shared__ __bf16 sX[2][BM][BK + PADH];   // 20,480 B (sync-staged; x is L2 hot)
  __shared__ float  sE[2][BN][BK + PADF];   // 18,432 B (async DMA)
  __shared__ float  sV[2][BN][BK + PADF];   // 18,432 B (async DMA)

  const int tid  = threadIdx.x;
  const int wave = tid >> 5, lane = tid & 31;
  const int wm = wave >> 1, wn = wave & 1;
  const int half = lane >> 4, r = lane & 15;
  const int tileM = blockIdx.x * BM;        // M inner -> weight tiles L2-shared
  const int tileN = blockIdx.y * BN;

  const float* xg = x    + (size_t)tileM * ACT;
  const float* eg = Wenc + (size_t)tileN * ACT;
  const float* vg = Wvar + (size_t)tileN * ACT;

  auto issueW = [&](int k0, int buf) {      // 2 x 512 b128 chunks, 2 per thread
    unsigned bE = lds_off(&sE[buf][0][0]);
    unsigned bV = lds_off(&sV[buf][0][0]);
    for (int c = tid; c < BN * BK / 4; c += 256) {
      int rr = c >> 3, cc = (c & 7) * 4;
      unsigned off = (unsigned)(rr * (BK + PADF) + cc) * 4u;
      async_b128(bE + off, eg + (size_t)rr * ACT + k0 + cc);
      async_b128(bV + off, vg + (size_t)rr * ACT + k0 + cc);
    }
  };
  auto stageX = [&](int k0, int buf) {      // float4 loads, cvt to bf16 in LDS
    for (int c = tid; c < BM * BK / 4; c += 256) {
      int rr = c >> 3, cc = (c & 7) * 4;
      float4 v4 = *(const float4*)(xg + (size_t)rr * ACT + k0 + cc);
      v4bf o;
      o[0] = (__bf16)v4.x; o[1] = (__bf16)v4.y;
      o[2] = (__bf16)v4.z; o[3] = (__bf16)v4.w;
      *(v4bf*)&sX[buf][rr][cc] = o;
    }
  };

  v8f accMu[2][2] = {};
  v8f accLv[2][2] = {};

  issueW(0, 0);
  stageX(0, 0);
  int buf = 0;
  for (int k0 = 0; k0 < ACT; k0 += BK, buf ^= 1) {
    wait_async0();          // my DMA into buf done
    __syncthreads();        // everyone's DMA/stores done; prior reads retired
    if (k0 + BK < ACT) {    // overlap next tile's DMA with this tile's WMMA
      issueW(k0 + BK, buf ^ 1);
      stageX(k0 + BK, buf ^ 1);
    }

    v16bf aF[2], eF[2], vF[2];
    #pragma unroll
    for (int mi = 0; mi < 2; ++mi)
      aF[mi] = frag_bf16(&sX[buf][wm * 32 + mi * 16 + r][0], half);
    #pragma unroll
    for (int ni = 0; ni < 2; ++ni) {
      int col = wn * 32 + ni * 16 + r;
      eF[ni] = frag_f32(&sE[buf][col][0], half);
      vF[ni] = frag_f32(&sV[buf][col][0], half);
    }
    #pragma unroll
    for (int mi = 0; mi < 2; ++mi)
      #pragma unroll
      for (int ni = 0; ni < 2; ++ni) {
        accMu[mi][ni] = __builtin_amdgcn_wmma_f32_16x16x32_bf16(
            false, aF[mi], false, eF[ni], (short)0, accMu[mi][ni], false, false);
        accLv[mi][ni] = __builtin_amdgcn_wmma_f32_16x16x32_bf16(
            false, aF[mi], false, vF[ni], (short)0, accLv[mi][ni], false, false);
      }
  }

  // epilogue: lanes 0-15: N=lane, M=v ; lanes 16-31: M=8+v
  #pragma unroll
  for (int mi = 0; mi < 2; ++mi)
    #pragma unroll
    for (int ni = 0; ni < 2; ++ni) {
      int col = tileN + wn * 32 + ni * 16 + r;
      float be = benc[col], bv = bvar[col];
      #pragma unroll
      for (int v = 0; v < 8; ++v) {
        int rowm = tileM + wm * 32 + mi * 16 + half * 8 + v;
        size_t idx = (size_t)rowm * DICT + col;
        mu[idx] = accMu[mi][ni][v] + be;
        lv[idx] = accLv[mi][ni][v] + bv;
      }
    }
}

// ----------------------------------------------- global top-k via histogram
__device__ __forceinline__ unsigned fkey(float f) {
  unsigned u = __float_as_uint(f);
  return (u & 0x80000000u) ? ~u : (u | 0x80000000u);   // order-preserving
}

__global__ void vsae_hist_zero(unsigned* __restrict__ hist) {
  int i = blockIdx.x * blockDim.x + threadIdx.x;
  if (i < 65537) hist[i] = 0;
}

__global__ void vsae_hist_build(const float4* __restrict__ mu4,
                                unsigned* __restrict__ hist) {
  const size_t n4 = (size_t)BATCH * DICT / 4;
  for (size_t i = (size_t)blockIdx.x * blockDim.x + threadIdx.x; i < n4;
       i += (size_t)gridDim.x * blockDim.x) {
    float4 m = mu4[i];
    atomicAdd(&hist[fkey(m.x) >> 16], 1u);
    atomicAdd(&hist[fkey(m.y) >> 16], 1u);
    atomicAdd(&hist[fkey(m.z) >> 16], 1u);
    atomicAdd(&hist[fkey(m.w) >> 16], 1u);
  }
}

__global__ void vsae_hist_scan(unsigned* __restrict__ hist) {
  if (threadIdx.x == 0 && blockIdx.x == 0) {
    unsigned long long cum = 0;
    int b = 65535;
    for (; b > 0; --b) {
      cum += hist[b];
      if (cum >= (unsigned long long)KTOTAL) break;
    }
    hist[65536] = ((unsigned)b) << 16;   // keep keys >= bin lower edge
  }
}

// --------------------------------------------- mask + reparameterize -------
// z = mu_m + eps*exp(0.5*clip(lv_m,-6,2)); emitted in bf16 for the decoder
__global__ void vsae_reparam(const float4* __restrict__ mu4,
                             const float4* __restrict__ lv4,
                             const float4* __restrict__ eps4,
                             const unsigned* __restrict__ thr,
                             v4bf* __restrict__ z4) {
  const unsigned tk = *thr;
  const size_t n4 = (size_t)BATCH * DICT / 4;
  for (size_t i = (size_t)blockIdx.x * blockDim.x + threadIdx.x; i < n4;
       i += (size_t)gridDim.x * blockDim.x) {
    float4 m = mu4[i], l = lv4[i], e = eps4[i];
    float mm[4] = {m.x, m.y, m.z, m.w};
    float ll[4] = {l.x, l.y, l.z, l.w};
    float ee[4] = {e.x, e.y, e.z, e.w};
    v4bf o;
    #pragma unroll
    for (int j = 0; j < 4; ++j) {
      bool keep = fkey(mm[j]) >= tk;
      float mv = keep ? mm[j] : 0.0f;
      float lc = keep ? ll[j] : 0.0f;
      lc = fminf(2.0f, fmaxf(-6.0f, lc));
      o[j] = (__bf16)(mv + ee[j] * __expf(0.5f * lc));
    }
    z4[i] = o;
  }
}

// ------------------------------------------------------------- decoder -----
// x_hat = z @ W_dec^T + b_dec ; z:[B,DICT] bf16, W_dec:[ACT,DICT] f32
__global__ __launch_bounds__(256)
void vsae_decoder_gemm(const __bf16* __restrict__ z,
                       const float* __restrict__ Wdec,
                       const float* __restrict__ bdec,
                       float* __restrict__ out)
{
  __shared__ __bf16 sZ[2][BM][BK + PADH];   // 20,480 B (async DMA, bf16)
  __shared__ float  sW[2][BN][BK + PADF];   // 18,432 B (async DMA, f32)

  const int tid  = threadIdx.x;
  const int wave = tid >> 5, lane = tid & 31;
  const int wm = wave >> 1, wn = wave & 1;
  const int half = lane >> 4, r = lane & 15;
  const int tileM = blockIdx.x * BM;
  const int tileN = blockIdx.y * BN;

  const __bf16* zg = z    + (size_t)tileM * DICT;
  const float*  wg = Wdec + (size_t)tileN * DICT;

  auto issue = [&](int k0, int buf) {
    unsigned bZ = lds_off(&sZ[buf][0][0]);
    for (int c = tid; c < BM * BK / 8; c += 256) {   // 512 bf16x8 chunks
      int rr = c >> 2, cc = (c & 3) * 8;
      async_b128(bZ + (unsigned)(rr * (BK + PADH) + cc) * 2u,
                 zg + (size_t)rr * DICT + k0 + cc);
    }
    unsigned bW = lds_off(&sW[buf][0][0]);
    for (int c = tid; c < BN * BK / 4; c += 256) {   // 512 f32x4 chunks
      int rr = c >> 3, cc = (c & 7) * 4;
      async_b128(bW + (unsigned)(rr * (BK + PADF) + cc) * 4u,
                 wg + (size_t)rr * DICT + k0 + cc);
    }
  };

  v8f acc[2][2] = {};

  issue(0, 0);
  int buf = 0;
  for (int k0 = 0; k0 < DICT; k0 += BK, buf ^= 1) {
    wait_async0();
    __syncthreads();
    if (k0 + BK < DICT) issue(k0 + BK, buf ^ 1);

    v16bf aF[2], bF[2];
    #pragma unroll
    for (int mi = 0; mi < 2; ++mi)
      aF[mi] = frag_bf16(&sZ[buf][wm * 32 + mi * 16 + r][0], half);
    #pragma unroll
    for (int ni = 0; ni < 2; ++ni)
      bF[ni] = frag_f32(&sW[buf][wn * 32 + ni * 16 + r][0], half);
    #pragma unroll
    for (int mi = 0; mi < 2; ++mi)
      #pragma unroll
      for (int ni = 0; ni < 2; ++ni)
        acc[mi][ni] = __builtin_amdgcn_wmma_f32_16x16x32_bf16(
            false, aF[mi], false, bF[ni], (short)0, acc[mi][ni], false, false);
  }

  #pragma unroll
  for (int mi = 0; mi < 2; ++mi)
    #pragma unroll
    for (int ni = 0; ni < 2; ++ni) {
      int col = tileN + wn * 32 + ni * 16 + r;
      float bd = bdec[col];
      #pragma unroll
      for (int v = 0; v < 8; ++v) {
        int rowm = tileM + wm * 32 + mi * 16 + half * 8 + v;
        out[(size_t)rowm * ACT + col] = acc[mi][ni][v] + bd;
      }
    }
}

// ---------------------------------------------------------------- launch ---
extern "C" void kernel_launch(void* const* d_in, const int* in_sizes, int n_in,
                              void* d_out, int out_size, void* d_ws, size_t ws_size,
                              hipStream_t stream) {
  const float* x    = (const float*)d_in[0];
  const float* eps  = (const float*)d_in[1];
  const float* Wenc = (const float*)d_in[2];
  const float* benc = (const float*)d_in[3];
  const float* Wvar = (const float*)d_in[4];
  const float* bvar = (const float*)d_in[5];
  const float* Wdec = (const float*)d_in[6];
  const float* bdec = (const float*)d_in[7];
  float* out = (float*)d_out;

  const size_t nMu = (size_t)BATCH * DICT;         // 67,108,864
  float*    mu   = (float*)d_ws;                   // [B, DICT] f32
  float*    lv   = mu + nMu;                       // [B, DICT] f32
  __bf16*   zb   = (__bf16*)(lv + nMu);            // [B, DICT] bf16
  unsigned* hist = (unsigned*)(zb + nMu);          // 65536 bins + threshold

  vsae_encoder_gemm<<<dim3(BATCH / BM, DICT / BN), 256, 0, stream>>>(
      x, Wenc, benc, Wvar, bvar, mu, lv);
  vsae_hist_zero<<<(65537 + 255) / 256, 256, 0, stream>>>(hist);
  vsae_hist_build<<<4096, 256, 0, stream>>>((const float4*)mu, hist);
  vsae_hist_scan<<<1, 32, 0, stream>>>(hist);
  vsae_reparam<<<8192, 256, 0, stream>>>(
      (const float4*)mu, (const float4*)lv, (const float4*)eps,
      hist + 65536, (v4bf*)zb);
  vsae_decoder_gemm<<<dim3(BATCH / BM, ACT / BN), 256, 0, stream>>>(
      zb, Wdec, bdec, out);
}
